// CrossAttentionBlock_10720238371216
// MI455X (gfx1250) — matrix operand order repaired and checked by reading, not compiled, not run on hardware
//
#include <hip/hip_runtime.h>
#include <hip/hip_bf16.h>

typedef __attribute__((ext_vector_type(16))) _Float16 v16h;
typedef __attribute__((ext_vector_type(8)))  _Float16 v8h;
typedef __attribute__((ext_vector_type(8)))  float    v8f;

#define B_   16
#define C_   512
#define L_   1024
#define NH_  8
#define HD_  64

union frag16 { v16h v; v8h h[2]; };

static __device__ __forceinline__ v8f wmma_f16(v16h a, v16h b, v8f c) {
    return __builtin_amdgcn_wmma_f32_16x16x32_f16(false, a, false, b, (short)0, c, false, false);
}

// A fragment (16x32 f16): lane holds row m=lane%16; halves 0-7 = K at 8*(lane/16),
// halves 8-15 = K at 16+8*(lane/16).  base -> element (m=0,k=0), row-major, leading dim ld.
static __device__ __forceinline__ v16h load_a_frag(const _Float16* base, int ld) {
    int lane = threadIdx.x & 31;
    int m  = lane & 15;
    int kb = (lane >> 4) * 8;
    const _Float16* p = base + (size_t)m * ld + kb;
    frag16 f;
    f.h[0] = *(const v8h*)(p);
    f.h[1] = *(const v8h*)(p + 16);
    return f.v;
}

// B fragment (32x16 f16): lane holds col n=lane%16; halves 0-7 = K at 8*(lane/16),
// halves 8-15 = K at 16+8*(lane/16).
// base -> element (k=0) of column n's storage (i.e. [N][K] layout), leading dim ld.
static __device__ __forceinline__ v16h load_b_frag(const _Float16* base, int ld) {
    int lane = threadIdx.x & 31;
    int n  = lane & 15;
    int kb = (lane >> 4) * 8;
    const _Float16* p = base + (size_t)n * ld + kb;
    frag16 f;
    f.h[0] = *(const v8h*)(p);
    f.h[1] = *(const v8h*)(p + 16);
    return f.v;
}

// ---------------- weight f32 -> f16 ----------------
__global__ void cvt_f32_f16(const float* __restrict__ in, _Float16* __restrict__ out, int n) {
    int i = blockIdx.x * 256 + threadIdx.x;
    if (i < n) out[i] = (_Float16)in[i];
}

// ---------------- GroupNorm + transpose to [B][L][C] f16 ----------------
// one block per (b, group); 32 groups of 16 channels x 1024 spatial
__global__ void groupnorm_t(const float* __restrict__ x, const float* __restrict__ w,
                            const float* __restrict__ bb, _Float16* __restrict__ out) {
    int b = blockIdx.x >> 5;
    int g = blockIdx.x & 31;
    const size_t base = ((size_t)b * C_ + g * 16) * L_;
    float s = 0.f, ss = 0.f;
    for (int i = threadIdx.x; i < 16 * L_; i += 256) {
        float v = x[base + i];
        s += v; ss += v * v;
    }
    #pragma unroll
    for (int off = 16; off; off >>= 1) {
        s  += __shfl_xor(s,  off, 32);
        ss += __shfl_xor(ss, off, 32);
    }
    __shared__ float red0[8], red1[8], stats[2];
    int wv = threadIdx.x >> 5;
    if ((threadIdx.x & 31) == 0) { red0[wv] = s; red1[wv] = ss; }
    __syncthreads();
    if (threadIdx.x == 0) {
        float S = 0.f, SS = 0.f;
        for (int i = 0; i < 8; ++i) { S += red0[i]; SS += red1[i]; }
        float mu  = S / 16384.f;
        float var = SS / 16384.f - mu * mu;
        stats[0] = mu;
        stats[1] = rsqrtf(var + 1e-5f);
    }
    __syncthreads();
    float mu = stats[0], rstd = stats[1];
    int cl = threadIdx.x & 15;          // channel within group
    int lt = threadIdx.x >> 4;          // spatial slice
    int c  = g * 16 + cl;
    float wc = w[c], bc = bb[c];
    for (int k = 0; k < 64; ++k) {
        int l = lt + 16 * k;
        float v = x[((size_t)b * C_ + c) * L_ + l];
        float r = (v - mu) * rstd * wc + bc;
        out[((size_t)b * L_ + l) * C_ + c] = (_Float16)r;   // coalesced over c
    }
}

// ---------------- conv1x1 GEMM:  D = W(MxK) * Xt^T  (+bias) ----------------
// Xt: [B][L][K] f16.  mode 0 (Q): out0[b][l][m] = val*scale (transposed b128 store)
// mode 1 (KV): m<512 -> Kt[b][l][m] ; m>=512 -> Vn[b][m-512][l] (natural, scalar)
__global__ void gemm_conv(const _Float16* __restrict__ W, const _Float16* __restrict__ Xt,
                          const float* __restrict__ bias, _Float16* __restrict__ out0,
                          _Float16* __restrict__ out1, int M, float scale, int mode) {
    int wid   = blockIdx.x * 4 + (threadIdx.x >> 5);
    int per_b = (M / 32) * (L_ / 64);
    int b  = wid / per_b;
    int rm = wid % per_b;
    int mt = rm / (L_ / 64);
    int nt = rm % (L_ / 64);
    int m0 = mt * 32, n0 = nt * 64;
    int lane = threadIdx.x & 31, nl = lane & 15, hf = lane >> 4;

    v8f acc[2][4] = {};
    for (int k0 = 0; k0 < C_; k0 += 32) {
        v16h a[2], bf[4];
        #pragma unroll
        for (int i = 0; i < 2; ++i)
            a[i] = load_a_frag(W + (size_t)(m0 + 16 * i) * C_ + k0, C_);
        #pragma unroll
        for (int j = 0; j < 4; ++j)
            bf[j] = load_b_frag(Xt + ((size_t)b * L_ + n0 + 16 * j) * C_ + k0, C_);
        #pragma unroll
        for (int i = 0; i < 2; ++i)
            #pragma unroll
            for (int j = 0; j < 4; ++j)
                acc[i][j] = wmma_f16(a[i], bf[j], acc[i][j]);
    }
    #pragma unroll
    for (int i = 0; i < 2; ++i) {
        int mb = m0 + 16 * i + 8 * hf;
        float bv[8];
        #pragma unroll
        for (int r = 0; r < 8; ++r) bv[r] = bias[mb + r];
        #pragma unroll
        for (int j = 0; j < 4; ++j) {
            int n = n0 + 16 * j + nl;
            if (mode == 0 || (m0 + 16 * i) < 512) {
                v8h pk;
                #pragma unroll
                for (int r = 0; r < 8; ++r)
                    pk[r] = (_Float16)((acc[i][j][r] + bv[r]) * scale);
                *(v8h*)(out0 + ((size_t)b * L_ + n) * C_ + mb) = pk;   // transposed
            } else {
                #pragma unroll
                for (int r = 0; r < 8; ++r)
                    out1[((size_t)b * C_ + (mb - 512 + r)) * L_ + n] =
                        (_Float16)(acc[i][j][r] + bv[r]);              // natural V
            }
        }
    }
}

// ---------------- flash attention: per (b, head, 16-row q tile) ----------------
__global__ void attn_flash(const _Float16* __restrict__ Qt, const _Float16* __restrict__ Kt,
                           const _Float16* __restrict__ Vn, _Float16* __restrict__ Ot) {
    __shared__ __align__(16) _Float16 Plds[4][16][32];
    int w    = threadIdx.x >> 5;
    int lane = threadIdx.x & 31;
    int wid  = blockIdx.x * 4 + w;
    int b  = wid >> 9;           // / (NH*64)
    int h  = (wid >> 6) & 7;
    int qt = wid & 63;
    int nl = lane & 15, hf = lane >> 4;

    v16h aq[2];
    const _Float16* Qb = Qt + ((size_t)b * L_ + qt * 16) * C_ + h * HD_;
    aq[0] = load_a_frag(Qb, C_);
    aq[1] = load_a_frag(Qb + 32, C_);

    v8f o[4] = {};
    float mr[8], lr[8];
    #pragma unroll
    for (int r = 0; r < 8; ++r) { mr[r] = -1e30f; lr[r] = 0.f; }

    for (int jt = 0; jt < 64; ++jt) {
        int j0 = jt * 32;
        v8f s0 = {}, s1 = {};
        const _Float16* Kb0 = Kt + ((size_t)b * L_ + j0) * C_ + h * HD_;
        const _Float16* Kb1 = Kb0 + (size_t)16 * C_;
        #pragma unroll
        for (int s = 0; s < 2; ++s) {
            v16h bk0 = load_b_frag(Kb0 + 32 * s, C_);
            v16h bk1 = load_b_frag(Kb1 + 32 * s, C_);
            s0 = wmma_f16(aq[s], bk0, s0);
            s1 = wmma_f16(aq[s], bk1, s1);
        }
        float p0[8], p1[8], alpha[8];
        #pragma unroll
        for (int r = 0; r < 8; ++r) {
            float v = fmaxf(s0[r], s1[r]);
            #pragma unroll
            for (int off = 1; off < 16; off <<= 1) v = fmaxf(v, __shfl_xor(v, off, 16));
            float nm = fmaxf(mr[r], v);
            alpha[r] = __expf(mr[r] - nm);
            mr[r] = nm;
            p0[r] = __expf(s0[r] - nm);
            p1[r] = __expf(s1[r] - nm);
            float rs = p0[r] + p1[r];
            #pragma unroll
            for (int off = 1; off < 16; off <<= 1) rs += __shfl_xor(rs, off, 16);
            lr[r] = lr[r] * alpha[r] + rs;
        }
        #pragma unroll
        for (int t = 0; t < 4; ++t)
            #pragma unroll
            for (int r = 0; r < 8; ++r) o[t][r] *= alpha[r];
        #pragma unroll
        for (int r = 0; r < 8; ++r) {
            Plds[w][r + 8 * hf][nl]      = (_Float16)p0[r];   // D-layout -> LDS
            Plds[w][r + 8 * hf][16 + nl] = (_Float16)p1[r];
        }
        __syncthreads();
        frag16 pf;
        {
            int kb = hf * 8;
            pf.h[0] = *(const v8h*)&Plds[w][nl][kb];          // A-fragment layout
            pf.h[1] = *(const v8h*)&Plds[w][nl][16 + kb];
        }
        #pragma unroll
        for (int t = 0; t < 4; ++t) {
            v16h bv = load_b_frag(Vn + ((size_t)b * C_ + h * HD_ + t * 16) * L_ + j0, L_);
            o[t] = wmma_f16(pf.v, bv, o[t]);
        }
        __syncthreads();
    }
    #pragma unroll
    for (int r = 0; r < 8; ++r) {
        float inv = 1.0f / lr[r];
        size_t row = (size_t)b * L_ + qt * 16 + r + 8 * hf;
        #pragma unroll
        for (int t = 0; t < 4; ++t)
            Ot[row * C_ + h * HD_ + t * 16 + nl] = (_Float16)(o[t][r] * inv);
    }
}

// ---------------- projection GEMM + bias + residual (f32 out) ----------------
__global__ void proj_res(const _Float16* __restrict__ Wp, const _Float16* __restrict__ Ot,
                         const float* __restrict__ bias, const float* __restrict__ xres,
                         float* __restrict__ out) {
    int wid = blockIdx.x * 4 + (threadIdx.x >> 5);
    int b  = wid >> 8;          // per_b = (512/32)*(1024/64) = 256
    int rm = wid & 255;
    int mt = rm >> 4, nt = rm & 15;
    int m0 = mt * 32, n0 = nt * 64;
    int lane = threadIdx.x & 31, nl = lane & 15, hf = lane >> 4;

    v8f acc[2][4] = {};
    for (int k0 = 0; k0 < C_; k0 += 32) {
        v16h a[2], bf[4];
        #pragma unroll
        for (int i = 0; i < 2; ++i)
            a[i] = load_a_frag(Wp + (size_t)(m0 + 16 * i) * C_ + k0, C_);
        #pragma unroll
        for (int j = 0; j < 4; ++j)
            bf[j] = load_b_frag(Ot + ((size_t)b * L_ + n0 + 16 * j) * C_ + k0, C_);
        #pragma unroll
        for (int i = 0; i < 2; ++i)
            #pragma unroll
            for (int j = 0; j < 4; ++j)
                acc[i][j] = wmma_f16(a[i], bf[j], acc[i][j]);
    }
    #pragma unroll
    for (int i = 0; i < 2; ++i) {
        int mb = m0 + 16 * i + 8 * hf;
        float bv[8];
        #pragma unroll
        for (int r = 0; r < 8; ++r) bv[r] = bias[mb + r];
        #pragma unroll
        for (int j = 0; j < 4; ++j) {
            int n = n0 + 16 * j + nl;
            #pragma unroll
            for (int r = 0; r < 8; ++r) {
                size_t idx = ((size_t)b * C_ + mb + r) * L_ + n;
                out[idx] = xres[idx] + acc[i][j][r] + bv[r];
            }
        }
    }
}

extern "C" void kernel_launch(void* const* d_in, const int* in_sizes, int n_in,
                              void* d_out, int out_size, void* d_ws, size_t ws_size,
                              hipStream_t stream) {
    const float* x        = (const float*)d_in[0];
    const float* ctx      = (const float*)d_in[1];
    const float* nq_w     = (const float*)d_in[2];
    const float* nq_b     = (const float*)d_in[3];
    const float* nkv_w    = (const float*)d_in[4];
    const float* nkv_b    = (const float*)d_in[5];
    const float* wq_f     = (const float*)d_in[6];
    const float* bq       = (const float*)d_in[7];
    const float* wkv_f    = (const float*)d_in[8];
    const float* bkv      = (const float*)d_in[9];
    const float* wp_f     = (const float*)d_in[10];
    const float* bp       = (const float*)d_in[11];

    char* ws = (char*)d_ws;
    const size_t SZ = (size_t)B_ * L_ * C_ * sizeof(_Float16);   // 16 MB
    _Float16* Xt  = (_Float16*)(ws);
    _Float16* Ct  = (_Float16*)(ws + SZ);
    _Float16* Qt  = (_Float16*)(ws + 2 * SZ);
    _Float16* Kt  = (_Float16*)(ws + 3 * SZ);
    _Float16* Vn  = (_Float16*)(ws + 4 * SZ);
    _Float16* Ot  = Xt;  // Xt is dead after the Q GEMM; safe to reuse
    _Float16* Wq  = (_Float16*)(ws + 5 * SZ);
    _Float16* Wkv = Wq + (size_t)C_ * C_;
    _Float16* Wp  = Wkv + (size_t)2 * C_ * C_;

    cvt_f32_f16<<<(C_ * C_ + 255) / 256, 256, 0, stream>>>(wq_f, Wq, C_ * C_);
    cvt_f32_f16<<<(2 * C_ * C_ + 255) / 256, 256, 0, stream>>>(wkv_f, Wkv, 2 * C_ * C_);
    cvt_f32_f16<<<(C_ * C_ + 255) / 256, 256, 0, stream>>>(wp_f, Wp, C_ * C_);

    groupnorm_t<<<B_ * 32, 256, 0, stream>>>(x,   nq_w,  nq_b,  Xt);
    groupnorm_t<<<B_ * 32, 256, 0, stream>>>(ctx, nkv_w, nkv_b, Ct);

    // Q = Wq*gn(x) scaled by hd^-0.5, stored transposed [B][L][C]
    gemm_conv<<<(B_ * (C_ / 32) * (L_ / 64)) / 4, 128, 0, stream>>>(
        Wq, Xt, bq, Qt, nullptr, C_, 0.125f, 0);
    // KV = Wkv*gn(ctx): K transposed [B][L][C], V natural [B][C][L]
    gemm_conv<<<(B_ * (2 * C_ / 32) * (L_ / 64)) / 4, 128, 0, stream>>>(
        Wkv, Ct, bkv, Kt, Vn, 2 * C_, 1.0f, 1);

    attn_flash<<<(B_ * NH_ * 64) / 4, 128, 0, stream>>>(Qt, Kt, Vn, Ot);

    proj_res<<<(B_ * (C_ / 32) * (L_ / 64)) / 4, 128, 0, stream>>>(
        Wp, Ot, bp, x, (float*)d_out);
}
